// BlocksCore_58153857188088
// MI455X (gfx1250) — matrix operand order repaired and checked by reading, not compiled, not run on hardware
//
#include <hip/hip_runtime.h>

// ---------------- problem constants (from reference) ----------------
#define B_    2048
#define NINP  512
#define NHID  2048
#define NB    8
#define BS    256
#define HI    4      // input heads
#define DKI   64
#define ATT   512
#define DVI   128
#define HC    4
#define DKC   32
#define DVC   32
#define G4    1024   // 4*BS gates

typedef __attribute__((ext_vector_type(16))) _Float16 v16h;
typedef __attribute__((ext_vector_type(8)))  _Float16 v8h;
typedef __attribute__((ext_vector_type(4)))  _Float16 v4h;
typedef __attribute__((ext_vector_type(8)))  float    v8f;

static __device__ __forceinline__ v8f vzero8() {
  v8f z = {0.f,0.f,0.f,0.f,0.f,0.f,0.f,0.f};
  return z;
}

static __device__ __forceinline__ float sigm(float x) {
  return 1.f / (1.f + __expf(-x));
}

static __device__ __forceinline__ v16h cat8(v8h lo, v8h hi) {
  return __builtin_shufflevector(lo, hi, 0,1,2,3,4,5,6,7,8,9,10,11,12,13,14,15);
}

// A fragment: 16x32 f16 from row-major f16 (leading dim lda, both 8-half aligned).
// lane<16: M=lane, K=k0+{0..7,16..23}; lane>=16: M=lane-16, K=k0+{8..15,24..31}
static __device__ __forceinline__ v16h load_a_frag16(const _Float16* __restrict__ A, int lda,
                                                     int k0, int lane) {
  int m  = lane & 15;
  int ko = k0 + ((lane >> 4) << 3);
  const _Float16* p = A + (size_t)m * lda + ko;
  v8h lo = *(const v8h*)p;
  v8h hi = *(const v8h*)(p + 16);
  return cat8(lo, hi);
}

// B fragment from W stored (N,K) row-major f16 (i.e. B = W^T): 2x16B, K=k0+{0..7,16..23} / {8..15,24..31}.
static __device__ __forceinline__ v16h load_b_frag16(const _Float16* __restrict__ W, int ldw,
                                                     int k0, int lane) {
  int n  = lane & 15;
  int kb = k0 + ((lane >> 4) << 3);
  const _Float16* p = W + (size_t)n * ldw + kb;
  v8h lo = *(const v8h*)p;
  v8h hi = *(const v8h*)(p + 16);
  return cat8(lo, hi);
}

static __device__ __forceinline__ v8f wmma_f16(v16h a, v16h b, v8f c) {
  return __builtin_amdgcn_wmma_f32_16x16x32_f16(false, a, false, b, (short)0, c, false, false);
}

// ======================= prep: f32 -> f16 convert (vec4) =========================
__global__ void cvt_f16_kernel(const float* __restrict__ in, _Float16* __restrict__ out,
                               int n4) {
  int i = blockIdx.x * blockDim.x + threadIdx.x;
  if (i >= n4) return;
  float4 v = ((const float4*)in)[i];
  v4h o;
  o[0] = (_Float16)v.x; o[1] = (_Float16)v.y; o[2] = (_Float16)v.z; o[3] = (_Float16)v.w;
  ((v4h*)out)[i] = o;
}

// ====== prep: batched transpose f32 (b,K,N) -> f16 (b,N,K), contiguous writes ====
__global__ void transpose_f16_kernel(const float* __restrict__ in, _Float16* __restrict__ out,
                                     int K, int N, int total) {
  int i = blockIdx.x * blockDim.x + threadIdx.x;
  if (i >= total) return;
  int nk = N * K;
  int b = i / nk, r = i - b * nk;
  int n = r / K, k = r - n * K;
  out[i] = (_Float16)in[(size_t)b * nk + (size_t)k * N + n];
}

// ========== generic per-block GEMM: C = A @ W^T, all-f16 operands ================
// blockIdx.y = block index n; per-n element offsets a_noff, w_noff, c_noff.
template <typename OutT>
__global__ void gemm_nk16_kernel(const _Float16* __restrict__ A, int lda, int a_noff,
                                 const _Float16* __restrict__ W, int ldw, int w_noff,
                                 OutT* __restrict__ C, int ldc, int c_noff,
                                 int mtiles, int ntiles, int K) {
  int lane = threadIdx.x & 31;
  int wv   = blockIdx.x * (blockDim.x >> 5) + (threadIdx.x >> 5);
  if (wv >= mtiles * ntiles) return;              // wave-uniform
  int n  = blockIdx.y;
  int mt = wv / ntiles, nt = wv % ntiles;
  const _Float16* Ab = A + (size_t)n * a_noff + (size_t)(mt * 16) * lda;
  const _Float16* Wb = W + (size_t)n * w_noff + (size_t)(nt * 16) * ldw;
  OutT*           Cb = C + (size_t)n * c_noff + (size_t)(mt * 16) * ldc + nt * 16;
  v8f acc = vzero8();
#pragma unroll 4
  for (int k0 = 0; k0 < K; k0 += 32) {
    acc = wmma_f16(load_a_frag16(Ab, lda, k0, lane),
                   load_b_frag16(Wb, ldw, k0, lane), acc);
  }
  int col = lane & 15;
  int rb  = (lane >> 4) << 3;
#pragma unroll
  for (int r = 0; r < 8; ++r)
    Cb[(size_t)(rb + r) * ldc + col] = (OutT)acc[r];
}

// ================= input attention: softmax over {null,input} ====================
// null key/value are zero => p_input = sigmoid(q.k/8), p_null = 1 - p_input
__global__ void inp_att_kernel(const float* __restrict__ inp, const _Float16* __restrict__ q,
                               const _Float16* __restrict__ k1, _Float16* __restrict__ inp_use,
                               float* __restrict__ iatt_null) {
  int b    = blockIdx.x;
  int n    = threadIdx.x >> 5;     // wave -> block
  int lane = threadIdx.x & 31;
  int h    = lane >> 3, sub = lane & 7;
  const _Float16* qp = q  + (size_t)b * NHID + n * BS + h * DKI + sub * 8;
  const _Float16* kp = k1 + (size_t)b * (HI * DKI) + h * DKI + sub * 8;
  float s = 0.f;
#pragma unroll
  for (int j = 0; j < 8; ++j) s += (float)qp[j] * (float)kp[j];
  s += __shfl_xor(s, 1); s += __shfl_xor(s, 2); s += __shfl_xor(s, 4);
  s *= 0.125f;                       // 1/sqrt(64)
  float p1 = sigm(s);
  float pn = 1.f - p1;
  const float* ip = inp + (size_t)b * NINP + h * DVI + sub * 16;
  _Float16* op = inp_use + (size_t)b * (NB * ATT) + n * ATT + h * DVI + sub * 16;
#pragma unroll
  for (int j = 0; j < 16; ++j) op[j] = (_Float16)(p1 * ip[j]);
  float v = (sub == 0) ? pn * 0.25f : 0.f;   // mean over 4 heads
  v += __shfl_xor(v, 8); v += __shfl_xor(v, 16);
  if (lane == 0) iatt_null[b * NB + n] = v;
}

// ================= top-(NB-TOPK)=2 null-attention blocks -> mask 0 ===============
__global__ void topk_mask_kernel(const float* __restrict__ iatt_null,
                                 float* __restrict__ mask_out) {
  int b = blockIdx.x * blockDim.x + threadIdx.x;
  if (b >= B_) return;
  float v[NB];
#pragma unroll
  for (int i = 0; i < NB; ++i) v[i] = iatt_null[b * NB + i];
  int i1 = 0; float m1 = v[0];
#pragma unroll
  for (int i = 1; i < NB; ++i) if (v[i] > m1) { m1 = v[i]; i1 = i; }
  int i2 = -1; float m2 = -3.0e38f;
#pragma unroll
  for (int i = 0; i < NB; ++i) if (i != i1 && v[i] > m2) { m2 = v[i]; i2 = i; }
#pragma unroll
  for (int i = 0; i < NB; ++i)
    mask_out[b * NB + i] = (i == i1 || i == i2) ? 0.f : 1.f;
}

// ===== fused block-diagonal LSTM: gates GEMMs (4 accum/wave) + nonlinearities ====
__global__ void lstm_gates_kernel(const _Float16* __restrict__ inp_use,
                                  const _Float16* __restrict__ hx16,
                                  const float* __restrict__ cx,
                                  const _Float16* __restrict__ w_ih, const _Float16* __restrict__ w_hh,
                                  const float* __restrict__ b_ih, const float* __restrict__ b_hh,
                                  float* __restrict__ hnew, float* __restrict__ cnew,
                                  _Float16* __restrict__ hnew16) {
  int lane = threadIdx.x & 31;
  int wv   = blockIdx.x * (blockDim.x >> 5) + (threadIdx.x >> 5);
  int n    = blockIdx.y;
  int mt   = wv >> 4;        // 128 batch tiles
  int jt   = wv & 15;        // 16 column groups within BS
  v8f acc[4];
#pragma unroll
  for (int g = 0; g < 4; ++g) acc[g] = vzero8();

  // gates += inp_use(512) @ w_ih[n]^T   (w_ih16 is (G4, ATT) = (N,K): contiguous)
  const _Float16* Aa = inp_use + (size_t)(mt * 16) * (NB * ATT) + n * ATT;
#pragma unroll 2
  for (int k0 = 0; k0 < ATT; k0 += 32) {
    v16h a = load_a_frag16(Aa, NB * ATT, k0, lane);
#pragma unroll
    for (int g = 0; g < 4; ++g) {
      const _Float16* Wb = w_ih + ((size_t)n * G4 + g * BS + jt * 16) * ATT;
      acc[g] = wmma_f16(a, load_b_frag16(Wb, ATT, k0, lane), acc[g]);
    }
  }
  // gates += h(256) @ w_hh[n]^T
  const _Float16* Ah = hx16 + (size_t)(mt * 16) * NHID + n * BS;
#pragma unroll 2
  for (int k0 = 0; k0 < BS; k0 += 32) {
    v16h a = load_a_frag16(Ah, NHID, k0, lane);
#pragma unroll
    for (int g = 0; g < 4; ++g) {
      const _Float16* Wb = w_hh + ((size_t)n * G4 + g * BS + jt * 16) * BS;
      acc[g] = wmma_f16(a, load_b_frag16(Wb, BS, k0, lane), acc[g]);
    }
  }
  int col = jt * 16 + (lane & 15);
  float bias[4];
#pragma unroll
  for (int g = 0; g < 4; ++g)
    bias[g] = b_ih[n * G4 + g * BS + col] + b_hh[n * G4 + g * BS + col];
  int rb = (lane >> 4) << 3;
#pragma unroll
  for (int r = 0; r < 8; ++r) {
    int row = mt * 16 + rb + r;
    size_t idx = (size_t)row * NHID + n * BS + col;
    float ig = acc[0][r] + bias[0];
    float fg = acc[1][r] + bias[1];
    float gg = acc[2][r] + bias[2];
    float og = acc[3][r] + bias[3];
    float cn = sigm(fg) * cx[idx] + sigm(ig) * tanhf(gg);
    float hn = sigm(og) * tanhf(cn);
    cnew[idx] = cn;
    hnew[idx] = hn;
    hnew16[idx] = (_Float16)hn;
  }
}

// ================= communication attention (8x8 blocks, 4 heads) =================
__global__ void comm_att_kernel(const _Float16* __restrict__ qc, const _Float16* __restrict__ kc,
                                const _Float16* __restrict__ vc, _Float16* __restrict__ oc) {
  __shared__ float qs[NB * HC * DKC];   // 1024
  __shared__ float ks[NB * HC * DKC];
  __shared__ float vs[NB * HC * DVC];
  __shared__ float sc[NB * NB * HC];    // 256 scores -> probs
  int b = blockIdx.x, t = threadIdx.x;
  size_t base = (size_t)b * (NB * HC * DKC);
#pragma unroll
  for (int j = 0; j < 4; ++j) {
    qs[t + j * 256] = (float)qc[base + t + j * 256];
    ks[t + j * 256] = (float)kc[base + t + j * 256];
    vs[t + j * 256] = (float)vc[base + t + j * 256];
  }
  __syncthreads();
  {
    int n = t >> 5, m = (t >> 2) & 7, hc = t & 3;
    const float* qp = qs + n * 128 + hc * 32;
    const float* kp = ks + m * 128 + hc * 32;
    float s = 0.f;
#pragma unroll
    for (int d = 0; d < 32; ++d) s += qp[d] * kp[d];
    sc[n * 32 + m * 4 + hc] = s * 0.17677669529663687f;  // 1/sqrt(32)
  }
  __syncthreads();
  if (t < 32) {
    int n = t >> 2, hc = t & 3;
    float mx = -3.0e38f;
#pragma unroll
    for (int m = 0; m < NB; ++m) mx = fmaxf(mx, sc[n * 32 + m * 4 + hc]);
    float e[NB]; float sum = 0.f;
#pragma unroll
    for (int m = 0; m < NB; ++m) { e[m] = __expf(sc[n * 32 + m * 4 + hc] - mx); sum += e[m]; }
    float inv = 1.f / sum;
#pragma unroll
    for (int m = 0; m < NB; ++m) sc[n * 32 + m * 4 + hc] = e[m] * inv;
  }
  __syncthreads();
#pragma unroll
  for (int j = 0; j < 4; ++j) {
    int idx = t + j * 256;
    int n = idx >> 7, hc = (idx >> 5) & 3, d = idx & 31;
    float o = 0.f;
#pragma unroll
    for (int m = 0; m < NB; ++m) o += sc[n * 32 + m * 4 + hc] * vs[m * 128 + hc * 32 + d];
    oc[base + idx] = (_Float16)o;
  }
}

// ==== fused final: oc @ {gate_c_w, fc_c_w}^T + gated tanh + residual + masked update
__global__ void comm_out_kernel(const _Float16* __restrict__ oc,
                                const _Float16* __restrict__ gate_wt, const float* __restrict__ gate_b,
                                const _Float16* __restrict__ fc_wt,   const float* __restrict__ fc_b,
                                const float* __restrict__ hnew,   const float* __restrict__ cnew,
                                const float* __restrict__ hx,     const float* __restrict__ cx,
                                const float* __restrict__ mask,   float* __restrict__ out) {
  int lane = threadIdx.x & 31;
  int wv   = blockIdx.x * (blockDim.x >> 5) + (threadIdx.x >> 5);
  int mt   = wv >> 4;     // 1024 row tiles over B*NB = 16384 rows
  int nt   = wv & 15;     // 16 col tiles over BS
  const _Float16* Aa = oc + (size_t)(mt * 16) * (HC * DVC);
  const _Float16* Wg = gate_wt + (size_t)(nt * 16) * (HC * DVC);
  const _Float16* Wf = fc_wt   + (size_t)(nt * 16) * (HC * DVC);
  v8f ag = vzero8(), af = vzero8();
#pragma unroll
  for (int k0 = 0; k0 < HC * DVC; k0 += 32) {
    v16h a = load_a_frag16(Aa, HC * DVC, k0, lane);
    ag = wmma_f16(a, load_b_frag16(Wg, HC * DVC, k0, lane), ag);
    af = wmma_f16(a, load_b_frag16(Wf, HC * DVC, k0, lane), af);
  }
  int col = nt * 16 + (lane & 15);
  float gb = gate_b[col], fb = fc_b[col];
  int rb = (lane >> 4) << 3;
#pragma unroll
  for (int r = 0; r < 8; ++r) {
    int R = mt * 16 + rb + r;            // flattened (b, n)
    int b = R >> 3, n = R & 7;
    size_t idx = (size_t)b * NHID + n * BS + col;
    float hn   = hnew[idx];
    float hatt = sigm(ag[r] + gb) * tanhf(af[r] + fb) + hn;  // residual +hg
    float hfin = hn + hatt;                                   // h_new + h_att
    float m    = mask[b * NB + n];
    out[idx]                      = m * hfin      + (1.f - m) * hx[idx];
    out[(size_t)B_ * NHID + idx]  = m * cnew[idx] + (1.f - m) * cx[idx];
  }
}

// =================================== launch ======================================
extern "C" void kernel_launch(void* const* d_in, const int* in_sizes, int n_in,
                              void* d_out, int out_size, void* d_ws, size_t ws_size,
                              hipStream_t stream) {
  (void)in_sizes; (void)n_in; (void)out_size; (void)ws_size;
  const float* inp    = (const float*)d_in[0];
  const float* hx     = (const float*)d_in[1];
  const float* cx     = (const float*)d_in[2];
  const float* wq_inp = (const float*)d_in[3];
  const float* wk_inp = (const float*)d_in[4];
  const float* wq_c   = (const float*)d_in[5];
  const float* wk_c   = (const float*)d_in[6];
  const float* wv_c   = (const float*)d_in[7];
  const float* fc_w   = (const float*)d_in[8];
  const float* fc_b   = (const float*)d_in[9];
  const float* gate_w = (const float*)d_in[10];
  const float* gate_b = (const float*)d_in[11];
  const float* w_ih   = (const float*)d_in[12];
  const float* w_hh   = (const float*)d_in[13];
  const float* b_ih   = (const float*)d_in[14];
  const float* b_hh   = (const float*)d_in[15];

  float* out = (float*)d_out;
  char*  ws  = (char*)d_ws;

  // ---------------- workspace layout (byte offsets, 64KB aligned) ----------------
  _Float16* hx16      = (_Float16*)(ws + 0);           //  8 MB
  _Float16* w_ih16    = (_Float16*)(ws + 8388608);     //  8 MB
  _Float16* w_hh16    = (_Float16*)(ws + 16777216);    //  4 MB
  _Float16* inp16     = (_Float16*)(ws + 20971520);    //  2 MB
  _Float16* wq_inp_t  = (_Float16*)(ws + 23068672);    //  1 MB
  _Float16* wk1_t     = (_Float16*)(ws + 24117248);    //  256 KB
  _Float16* wqct      = (_Float16*)(ws + 24379392);    //  512 KB
  _Float16* wkct      = (_Float16*)(ws + 24903680);    //  512 KB
  _Float16* wvct      = (_Float16*)(ws + 25427968);    //  512 KB
  _Float16* gate_wt   = (_Float16*)(ws + 25952256);    //  64 KB
  _Float16* fc_wt     = (_Float16*)(ws + 26017792);    //  64 KB
  float*    iatt      = (float*)   (ws + 26083328);    //  64 KB
  float*    hnew32    = (float*)   (ws + 26148864);    // 16 MB
  float*    cnew32    = (float*)   (ws + 42926080);    // 16 MB
  _Float16* hnew16    = (_Float16*)(ws + 59703296);    //  8 MB
  _Float16* inp_use16 = (_Float16*)(ws + 68091904);    // 16 MB (reused below)
  _Float16* qc16      = (_Float16*)(ws + 68091904);    //  4 MB (after LSTM)
  _Float16* kc16      = (_Float16*)(ws + 72286208);    //  4 MB
  _Float16* vc16      = (_Float16*)(ws + 76480512);    //  4 MB
  _Float16* oc16      = (_Float16*)(ws + 80674816);    //  4 MB
  _Float16* k1_16     = (_Float16*)(ws + 84869120);    //  1 MB
  _Float16* q16       = (_Float16*)(ws + 85917696);    //  8 MB   (peak ~94 MB)

  float* maskp = out + (size_t)2 * B_ * NHID;          // third output

  dim3 blk(256);   // 8 waves/WG

  // ---- prep: f16 conversions ----
  cvt_f16_kernel<<<dim3(4096), blk, 0, stream>>>(hx,   hx16,   (B_ * NHID) / 4);
  cvt_f16_kernel<<<dim3(1024), blk, 0, stream>>>(inp,  inp16,  (B_ * NINP) / 4);
  cvt_f16_kernel<<<dim3(4096), blk, 0, stream>>>(w_ih, w_ih16, (NB * G4 * ATT) / 4);
  cvt_f16_kernel<<<dim3(2048), blk, 0, stream>>>(w_hh, w_hh16, (NB * G4 * BS) / 4);
  // ---- prep: (K,N)->(N,K) f16 transposes ----
  transpose_f16_kernel<<<dim3(512),  blk, 0, stream>>>(wk_inp + (size_t)NINP * (HI * DKI),
                                                       wk1_t, NINP, HI * DKI, NINP * HI * DKI);
  transpose_f16_kernel<<<dim3(2048), blk, 0, stream>>>(wq_inp, wq_inp_t, BS, HI * DKI,
                                                       NB * BS * HI * DKI);
  transpose_f16_kernel<<<dim3(1024), blk, 0, stream>>>(wq_c, wqct, BS, HC * DKC, NB * BS * HC * DKC);
  transpose_f16_kernel<<<dim3(1024), blk, 0, stream>>>(wk_c, wkct, BS, HC * DKC, NB * BS * HC * DKC);
  transpose_f16_kernel<<<dim3(1024), blk, 0, stream>>>(wv_c, wvct, BS, HC * DVC, NB * BS * HC * DVC);
  transpose_f16_kernel<<<dim3(128),  blk, 0, stream>>>(gate_w, gate_wt, HC * DVC, BS, HC * DVC * BS);
  transpose_f16_kernel<<<dim3(128),  blk, 0, stream>>>(fc_w,   fc_wt,   HC * DVC, BS, HC * DVC * BS);

  // k1 = inp @ wk_inp[1]     (M=2048, N=256, K=512)
  gemm_nk16_kernel<_Float16><<<dim3(256, 1), blk, 0, stream>>>(
      inp16, NINP, 0, wk1_t, NINP, 0, k1_16, HI * DKI, 0, 128, 16, NINP);
  // q[n] = hb[n] @ wq_inp[n] (per n: M=2048, N=256, K=256)
  gemm_nk16_kernel<_Float16><<<dim3(256, NB), blk, 0, stream>>>(
      hx16, NHID, BS, wq_inp_t, BS, BS * (HI * DKI), q16, NHID, BS, 128, 16, BS);
  // input attention -> inp_use16, iatt_null
  inp_att_kernel<<<dim3(B_), blk, 0, stream>>>(inp, q16, k1_16, inp_use16, iatt);
  // top-2 null blocks -> mask (written straight to d_out tail)
  topk_mask_kernel<<<dim3(B_ / 256), blk, 0, stream>>>(iatt, maskp);
  // fused block LSTM
  lstm_gates_kernel<<<dim3(256, NB), blk, 0, stream>>>(
      inp_use16, hx16, cx, w_ih16, w_hh16, b_ih, b_hh, hnew32, cnew32, hnew16);
  // comm q/k/v projections (per n: M=2048, N=128, K=256)
  gemm_nk16_kernel<_Float16><<<dim3(128, NB), blk, 0, stream>>>(
      hnew16, NHID, BS, wqct, BS, (HC * DKC) * BS, qc16, NB * HC * DKC, HC * DKC, 128, 8, BS);
  gemm_nk16_kernel<_Float16><<<dim3(128, NB), blk, 0, stream>>>(
      hnew16, NHID, BS, wkct, BS, (HC * DKC) * BS, kc16, NB * HC * DKC, HC * DKC, 128, 8, BS);
  gemm_nk16_kernel<_Float16><<<dim3(128, NB), blk, 0, stream>>>(
      hnew16, NHID, BS, wvct, BS, (HC * DVC) * BS, vc16, NB * HC * DVC, HC * DVC, 128, 8, BS);
  // 8x8 block attention, 4 heads
  comm_att_kernel<<<dim3(B_), blk, 0, stream>>>(qc16, kc16, vc16, oc16);
  // fused output projection + gated tanh + residual + masked state update
  comm_out_kernel<<<dim3(2048), blk, 0, stream>>>(
      oc16, gate_wt, gate_b, fc_wt, fc_b, hnew32, cnew32, hx, cx, maskp, out);
}
